// OutProductMean_18657337934597
// MI455X (gfx1250) — matrix operand order repaired and checked by reading, not compiled, not run on hardware
//
#include <hip/hip_runtime.h>

typedef float v8f  __attribute__((ext_vector_type(8)));
typedef __bf16 v16bf __attribute__((ext_vector_type(16)));
typedef __bf16 bf16x8 __attribute__((ext_vector_type(8)));
typedef unsigned short u16;

#define N_RES 384
#define S_DEPTH 128
#define C_FEAT 64
#define P_PROJ 32
#define CZ_OUT 128
#define MP (N_RES * P_PROJ)      // 12288 flattened (i*32+d)
#define AB_STR 40                // bf16 elems per LDS row (80B, 16B-aligned, conflict-free)
#define O_STR 1032               // bf16 elems per sO row (2064B = 16*129, conflict-free)
#define STG (4 * 128 * AB_STR)   // u16 elems per stage buffer (Ah|Al|Bh|Bl)
#define ARR (128 * AB_STR)       // u16 elems per array within a stage

// ---------------- bf16 split helpers (pure bit ops, RNE) -------------------
__device__ __forceinline__ u16 f32_to_bf16(float x) {
    unsigned int u = __float_as_uint(x);
    unsigned int r = u + 0x7FFFu + ((u >> 16) & 1u);
    return (u16)(r >> 16);
}
__device__ __forceinline__ float bf16_to_f32(u16 h) {
    return __uint_as_float(((unsigned int)h) << 16);
}
__device__ __forceinline__ void split_bf16(float x, u16& hi, u16& lo) {
    hi = f32_to_bf16(x);
    lo = f32_to_bf16(x - bf16_to_f32(hi));
}

// ---------------- CDNA5 async global->LDS copy (ASYNCcnt path) -------------
// Per-lane: LDS[vdst] = MEM[vaddr], 16B. Generic LDS pointers keep the LDS
// byte offset in their low 32 bits (ISA 10.2 aperture mapping).
__device__ __forceinline__ void async_b128(const u16* lds_dst, const void* gsrc) {
    unsigned lds = (unsigned)(unsigned long long)lds_dst;
    asm volatile("global_load_async_to_lds_b128 %0, %1, off"
                 :: "v"(lds), "v"(gsrc) : "memory");
}
__device__ __forceinline__ void wait_async0() {
    asm volatile("s_wait_asynccnt 0" ::: "memory");
}

// ---------------- WMMA fragment loaders (contiguous b128 pairs) ------------
// A-matrix 16x32 bf16: lane holds K {kb..kb+7, kb+16..kb+23}, kb=(lane>>4)*8
__device__ __forceinline__ v16bf ld_fragA(const u16* p) {
    bf16x8 c0 = *(const bf16x8*)(p);
    bf16x8 c1 = *(const bf16x8*)(p + 16);
    return __builtin_shufflevector(c0, c1, 0,1,2,3,4,5,6,7,8,9,10,11,12,13,14,15);
}
// B-matrix 32x16 bf16: lane n holds K {kb..kb+7, kb+16..kb+23}, kb=(lane>>4)*8
__device__ __forceinline__ v16bf ld_fragB(const u16* p) {
    bf16x8 c0 = *(const bf16x8*)(p);
    bf16x8 c1 = *(const bf16x8*)(p + 16);
    return __builtin_shufflevector(c0, c1, 0,1,2,3,4,5,6,7,8,9,10,11,12,13,14,15);
}
__device__ __forceinline__ v8f wmma_bf16(v16bf a, v16bf b, v8f c) {
    return __builtin_amdgcn_wmma_f32_16x16x32_bf16(false, a, false, b, (short)0, c, false, false);
}

// ---------------------------------------------------------------------------
// Kernel 1: LayerNorm + dual projection + mask, split-bf16 output.
// One wave per (s,i) row. Output layout [m = i*32+d][s] (K-contiguous rows
// so GEMM1 staging is pure b128 traffic; L2 absorbs the strided b16 stores).
// ---------------------------------------------------------------------------
__global__ __launch_bounds__(256) void opm_ln_proj_kernel(
    const float* __restrict__ M, const float* __restrict__ Mmask,
    const float* __restrict__ ln_g, const float* __restrict__ ln_b,
    const float* __restrict__ Wa, const float* __restrict__ ba,
    const float* __restrict__ Wb, const float* __restrict__ bb,
    u16* __restrict__ Lhi, u16* __restrict__ Llo,
    u16* __restrict__ Rhi, u16* __restrict__ Rlo)
{
    __shared__ float sWa[P_PROJ * 65];
    __shared__ float sWb[P_PROJ * 65];
    __shared__ float sMn[8 * C_FEAT];

    const int t = threadIdx.x;
    for (int idx = t; idx < P_PROJ * C_FEAT; idx += 256) {
        const int d = idx >> 6, c = idx & 63;
        sWa[d * 65 + c] = Wa[idx];
        sWb[d * 65 + c] = Wb[idx];
    }
    __syncthreads();

    const int wave = t >> 5, lane = t & 31;
    const int row  = blockIdx.x * 8 + wave;        // row = s*384 + i
    const float* mrow = M + (size_t)row * C_FEAT;

    const float x0 = mrow[lane];
    const float x1 = mrow[lane + 32];
    float s1 = x0 + x1, s2 = x0 * x0 + x1 * x1;
    #pragma unroll
    for (int off = 16; off > 0; off >>= 1) {
        s1 += __shfl_xor(s1, off, 32);
        s2 += __shfl_xor(s2, off, 32);
    }
    const float mu   = s1 * (1.0f / 64.0f);
    const float var  = s2 * (1.0f / 64.0f) - mu * mu;
    const float rstd = rsqrtf(var + 1e-5f);

    sMn[wave * 64 + lane]      = (x0 - mu) * rstd * ln_g[lane]      + ln_b[lane];
    sMn[wave * 64 + lane + 32] = (x1 - mu) * rstd * ln_g[lane + 32] + ln_b[lane + 32];
    __syncthreads();

    float accA = ba[lane], accB = bb[lane];
    const float* mn = &sMn[wave * 64];
    #pragma unroll 8
    for (int c = 0; c < C_FEAT; ++c) {
        const float v = mn[c];
        accA += v * sWa[lane * 65 + c];
        accB += v * sWb[lane * 65 + c];
    }
    const float mk = Mmask[row];
    const float la = mk * accA, ra = mk * accB;

    const int s_idx = row / N_RES;
    const int gi    = row - s_idx * N_RES;
    const size_t o  = ((size_t)gi * P_PROJ + lane) * S_DEPTH + s_idx;  // [m][s]
    u16 h, l;
    split_bf16(la, h, l); Lhi[o] = h; Llo[o] = l;
    split_bf16(ra, h, l); Rhi[o] = h; Rlo[o] = l;
}

// ---------------------------------------------------------------------------
// Kernel 1b: split Wo into hi/lo bf16 (row-major [o][de], B-frag-ready)
// ---------------------------------------------------------------------------
__global__ __launch_bounds__(256) void opm_split_wo_kernel(
    const float* __restrict__ Wo, u16* __restrict__ WoHi, u16* __restrict__ WoLo)
{
    const int idx = blockIdx.x * 256 + threadIdx.x;     // 131072 elems
    u16 h, l;
    split_bf16(Wo[idx], h, l);
    WoHi[idx] = h; WoLo[idx] = l;
}

// ---------------------------------------------------------------------------
// Kernel 2: fused O = L^T R (K=s=128) and Z = O * Wo^T (K=1024), split-bf16
// 3-pass WMMA (Ah*Bh + Ah*Bl + Al*Bh) with f32 accumulators.
// Staging via double-buffered GLOBAL_LOAD_ASYNC_TO_LDS_B128: chunk kc+1 DMA
// overlaps chunk kc's 24 WMMAs; drained with s_wait_asynccnt before barrier.
// ---------------------------------------------------------------------------
__global__ __launch_bounds__(256) void opm_gemm_kernel(
    const u16* __restrict__ Lhi, const u16* __restrict__ Llo,
    const u16* __restrict__ Rhi, const u16* __restrict__ Rlo,
    const u16* __restrict__ WoHi, const u16* __restrict__ WoLo,
    const float* __restrict__ Mmask, const float* __restrict__ bo,
    float* __restrict__ Z)
{
    extern __shared__ char smem[];
    u16* sStage = (u16*)smem;                 // [2][Ah|Al|Bh|Bl][128*AB_STR]
    u16* sOh    = sStage + 2 * STG;           // [16][O_STR]
    u16* sOl    = sOh + 16 * O_STR;
    float* sNorm = (float*)(sOl + 16 * O_STR);

    const int t    = threadIdx.x;
    const int wave = t >> 5;
    const int lane = t & 31;
    const int l15  = lane & 15;
    const int jb   = blockIdx.x;
    const int ib   = blockIdx.y;

    if (t < 16) {
        const int gi = ib * 4 + (t >> 2);
        const int gj = jb * 4 + (t & 3);
        float acc = 0.0f;
        for (int s = 0; s < S_DEPTH; ++s)
            acc += Mmask[s * N_RES + gi] * Mmask[s * N_RES + gj];
        sNorm[t] = 1.0f / (0.001f + acc);
    }

    const v8f vzero = {0.f, 0.f, 0.f, 0.f, 0.f, 0.f, 0.f, 0.f};
    v8f acc[8];
    #pragma unroll
    for (int tn = 0; tn < 8; ++tn) acc[tn] = vzero;

    const int mbase = ib * 128;
    const int nbase = jb * 128;
    const int aoff  = (lane >> 4) * 8;        // A-frag lane K offset
    const int boff  = (lane >> 4) * 8;        // B-frag lane K offset

    // GEMM2 row pointers (defined early so we can prefetch Wo into GL2
    // while GEMM1 runs)
    const int o = wave * 16 + l15;
    const u16* wRowH = WoHi + (size_t)o * 1024;
    const u16* wRowL = WoLo + (size_t)o * 1024;
    __builtin_prefetch(wRowH, 0, 2);
    __builtin_prefetch(wRowH + 512, 0, 2);
    __builtin_prefetch(wRowL, 0, 2);
    __builtin_prefetch(wRowL + 512, 0, 2);

    // staging assignment: thread copies 16 bf16 (2 x async b128) per array
    const int cm = t & 127;                   // row (m or n) to copy
    const int cp = (t >> 7) * 16;             // element offset within 32-wide row
    const int ldst = cm * AB_STR + cp;

    auto issue_stage = [&](int kc, int sel) {
        const size_t gA = ((size_t)(mbase + cm) * S_DEPTH) + kc * 32 + cp;
        const size_t gB = ((size_t)(nbase + cm) * S_DEPTH) + kc * 32 + cp;
        u16* Ah = sStage + sel * STG + ldst;
        u16* Al = Ah + ARR;
        u16* Bh = Al + ARR;
        u16* Bl = Bh + ARR;
        async_b128(Ah,     Lhi + gA);
        async_b128(Ah + 8, Lhi + gA + 8);
        async_b128(Al,     Llo + gA);
        async_b128(Al + 8, Llo + gA + 8);
        async_b128(Bh,     Rhi + gB);
        async_b128(Bh + 8, Rhi + gB + 8);
        async_b128(Bl,     Rlo + gB);
        async_b128(Bl + 8, Rlo + gB + 8);
    };

    // ---------------- GEMM1: O[m][n] = sum_s L[m][s] * R[n][s] -------------
    issue_stage(0, 0);
    wait_async0();
    __syncthreads();

    for (int kc = 0; kc < 4; ++kc) {
        const int cur = kc & 1;
        if (kc < 3) issue_stage(kc + 1, 1 - cur);   // overlaps with WMMAs below

        const u16* cAh = sStage + cur * STG;
        const u16* cAl = cAh + ARR;
        const u16* cBh = cAl + ARR;
        const u16* cBl = cBh + ARR;

        const int mrow = wave * 16 + l15;
        const v16bf Ah = ld_fragA(cAh + mrow * AB_STR + aoff);
        const v16bf Al = ld_fragA(cAl + mrow * AB_STR + aoff);
        #pragma unroll
        for (int tn = 0; tn < 8; ++tn) {
            const int n = tn * 16 + l15;
            const v16bf Bh = ld_fragB(cBh + n * AB_STR + boff);
            const v16bf Bl = ld_fragB(cBl + n * AB_STR + boff);
            acc[tn] = wmma_bf16(Ah, Bh, acc[tn]);
            acc[tn] = wmma_bf16(Ah, Bl, acc[tn]);
            acc[tn] = wmma_bf16(Al, Bh, acc[tn]);
        }
        wait_async0();
        __syncthreads();
    }

    // ---------------- spill O chunk to LDS as split-bf16 [pair][d*32+e] ----
    {
        const int half = lane >> 4;
        #pragma unroll
        for (int tn = 0; tn < 8; ++tn) {
            const int n  = tn * 16 + l15;
            const int jj = n >> 5, e = n & 31;
            #pragma unroll
            for (int r = 0; r < 8; ++r) {
                const int m  = wave * 16 + r + half * 8;
                const int ii = m >> 5, d = m & 31;
                const int oi = (ii * 4 + jj) * O_STR + d * 32 + e;
                u16 h, l;
                split_bf16(acc[tn][r], h, l);
                sOh[oi] = h; sOl[oi] = l;
            }
        }
    }
    __syncthreads();

    // ---------------- GEMM2: Z[pair][o] = O[pair][de] . Wo[o][de] ----------
    v8f zacc = vzero;
    const u16* oRowH = sOh + l15 * O_STR;
    const u16* oRowL = sOl + l15 * O_STR;
    for (int k0 = 0; k0 < 1024; k0 += 32) {
        const v16bf Ah = ld_fragA(oRowH + k0 + aoff);
        const v16bf Al = ld_fragA(oRowL + k0 + aoff);
        const v16bf Bh = ld_fragB(wRowH + k0 + boff);
        const v16bf Bl = ld_fragB(wRowL + k0 + boff);
        zacc = wmma_bf16(Ah, Bh, zacc);
        zacc = wmma_bf16(Ah, Bl, zacc);
        zacc = wmma_bf16(Al, Bh, zacc);
    }

    // ---------------- epilogue: +bo, * 1/(0.001+norm), store ---------------
    const float bias = bo[o];
    const int half = lane >> 4;
    #pragma unroll
    for (int r = 0; r < 8; ++r) {
        const int pair = r + half * 8;
        const int gi = ib * 4 + (pair >> 2);
        const int gj = jb * 4 + (pair & 3);
        Z[((size_t)gi * N_RES + gj) * CZ_OUT + o] = (zacc[r] + bias) * sNorm[pair];
    }
}

extern "C" void kernel_launch(void* const* d_in, const int* in_sizes, int n_in,
                              void* d_out, int out_size, void* d_ws, size_t ws_size,
                              hipStream_t stream) {
    const float* M    = (const float*)d_in[0];
    const float* mask = (const float*)d_in[1];
    const float* ln_g = (const float*)d_in[2];
    const float* ln_b = (const float*)d_in[3];
    const float* Wa   = (const float*)d_in[4];
    const float* ba   = (const float*)d_in[5];
    const float* Wb   = (const float*)d_in[6];
    const float* bb   = (const float*)d_in[7];
    const float* Wo   = (const float*)d_in[8];
    const float* bo   = (const float*)d_in[9];
    float* Z = (float*)d_out;

    const size_t nLR = (size_t)MP * S_DEPTH;        // 1572864 elems
    u16* Lhi  = (u16*)d_ws;
    u16* Llo  = Lhi + nLR;
    u16* Rhi  = Llo + nLR;
    u16* Rlo  = Rhi + nLR;
    u16* WoHi = Rlo + nLR;                          // 131072 elems
    u16* WoLo = WoHi + (size_t)CZ_OUT * 1024;

    opm_ln_proj_kernel<<<dim3(S_DEPTH * N_RES / 8), dim3(256), 0, stream>>>(
        M, mask, ln_g, ln_b, Wa, ba, Wb, bb, Lhi, Llo, Rhi, Rlo);

    opm_split_wo_kernel<<<dim3(CZ_OUT * 1024 / 256), dim3(256), 0, stream>>>(
        Wo, WoHi, WoLo);

    const size_t lds_bytes = (size_t)(2 * STG + 2 * 16 * O_STR) * sizeof(u16)
                           + 16 * sizeof(float);
    opm_gemm_kernel<<<dim3(N_RES / 4, N_RES / 4), dim3(256), lds_bytes, stream>>>(
        Lhi, Llo, Rhi, Rlo, WoHi, WoLo, mask, bo, Z);
}
